// SparseGraphAttentionLayer_83476984365497
// MI455X (gfx1250) — matrix operand order repaired and checked by reading, not compiled, not run on hardware
//
#include <hip/hip_runtime.h>

#define NF 256

typedef __attribute__((ext_vector_type(16))) __bf16 v16bf;
typedef __attribute__((ext_vector_type(8)))  float  v8f;

union ABfrag { unsigned u[8]; v16bf v; };

__device__ __forceinline__ unsigned short f2bf(float f) {
    unsigned u = __float_as_uint(f);
    unsigned r = u + 0x7fffu + ((u >> 16) & 1u);   // round-to-nearest-even
    return (unsigned short)(r >> 16);
}

// Monotone float <-> signed-int encoding (involution); int compare == float compare.
__device__ __forceinline__ int encf(float f) {
    int i = __float_as_int(f);
    return i < 0 ? (i ^ 0x7fffffff) : i;
}
__device__ __forceinline__ float decf(int i) {
    return __int_as_float(i < 0 ? (i ^ 0x7fffffff) : i);
}

// ---------------------------------------------------------------------------
// K0: convert W [k][n] f32 -> Wt [n][k] bf16 ; init segment-max (enc -inf) & denom
// ---------------------------------------------------------------------------
__global__ void k_init(const float* __restrict__ W, unsigned short* __restrict__ Wt,
                       int* __restrict__ menc, float* __restrict__ denom, int N) {
    int idx = blockIdx.x * blockDim.x + threadIdx.x;
    if (idx < NF * NF) {
        int k = idx >> 8, n = idx & 255;
        Wt[n * NF + k] = f2bf(W[k * NF + n]);
    }
    if (idx < N) {
        menc[idx]  = (int)0x80000000;   // <= enc(any finite float)
        denom[idx] = 0.0f;
    }
}

// K0b: zero the output accumulator (d_out is poisoned / holds stale results)
__global__ void k_zero(float* __restrict__ out, int n) {
    int idx = blockIdx.x * blockDim.x + threadIdx.x;
    if (idx < n) out[idx] = 0.0f;
}

// ---------------------------------------------------------------------------
// K1: Wh = x @ W + b  via bf16 WMMA, fp32 accumulate.
// 3125 blocks x 512 threads (16 waves); block = 16 M-rows, wave = one 16x16 tile.
// ---------------------------------------------------------------------------
__global__ __launch_bounds__(512) void k_gemm(const float* __restrict__ x,
                                              const unsigned* __restrict__ Wtu,
                                              const float* __restrict__ Wb,
                                              float* __restrict__ Wh) {
    __shared__ unsigned ldsA[16 * 128];   // 16 rows x 256 K, bf16 pairs (8 KB)

    const int tid = threadIdx.x;
    const int m0  = blockIdx.x * 16;

    // Stage x tile -> LDS as bf16 (each thread converts 8 consecutive floats)
    {
        const int base = tid * 8;                 // element idx in [0, 4096)
        const int r = base >> 8;
        const int c = base & 255;
        const float* src = x + (size_t)(m0 + r) * NF + c;
#pragma unroll
        for (int j = 0; j < 4; ++j) {
            float f0 = src[2 * j], f1 = src[2 * j + 1];
            ldsA[(base >> 1) + j] =
                (unsigned)f2bf(f0) | ((unsigned)f2bf(f1) << 16);
        }
    }
    __syncthreads();

    const int wave = tid >> 5;
    const int lane = tid & 31;
    const int half = lane >> 4;     // 0: lanes 0-15, 1: lanes 16-31
    const int lm   = lane & 15;
    const int n    = wave * 16 + lm;

    v8f acc = {0.f, 0.f, 0.f, 0.f, 0.f, 0.f, 0.f, 0.f};

#pragma unroll
    for (int s = 0; s < 8; ++s) {
        const int kb = s * 32;
        ABfrag a, b;
        // A fragment: 16-bit A 16x32 layout (ISA 7.12.2)
#pragma unroll
        for (int v = 0; v < 4; ++v) {
            int k = kb + 2 * v + 8 * half;
            a.u[v] = ldsA[lm * 128 + (k >> 1)];
        }
#pragma unroll
        for (int v = 0; v < 4; ++v) {
            int k = kb + 16 + 2 * v + 8 * half;
            a.u[4 + v] = ldsA[lm * 128 + (k >> 1)];
        }
        // B fragment: 32x16, K = 2(v&3) + 8*half + 16*(v>>2), column n (Wt is [n][k] bf16)
#pragma unroll
        for (int v = 0; v < 8; ++v) {
            int k = kb + 2 * (v & 3) + 8 * half + 16 * (v >> 2);
            b.u[v] = Wtu[n * 128 + (k >> 1)];
        }
        acc = __builtin_amdgcn_wmma_f32_16x16x32_bf16(
            false, a.v, false, b.v, (short)0, acc, false, false);
    }

    const float bias = Wb[n];
#pragma unroll
    for (int r = 0; r < 8; ++r) {
        int m = r + 8 * half;                    // C/D layout: VGPR r -> M = r + 8*half
        Wh[(size_t)(m0 + m) * NF + n] = acc[r] + bias;
    }
}

// ---------------------------------------------------------------------------
// K2: f1[n] = Wh[n].a_w[0:256], f2[n] = Wh[n].a_w[256:512]  (one wave32/node)
// ---------------------------------------------------------------------------
__global__ __launch_bounds__(256) void k_gemv(const float* __restrict__ Wh,
                                              const float* __restrict__ aw,
                                              float* __restrict__ f1,
                                              float* __restrict__ f2, int N) {
    int gid  = blockIdx.x * blockDim.x + threadIdx.x;
    int node = gid >> 5;
    int lane = gid & 31;
    if (node >= N) return;
    const float* w = Wh + (size_t)node * NF;
    float s1 = 0.f, s2 = 0.f;
#pragma unroll
    for (int j = 0; j < 8; ++j) {
        int f   = lane + 32 * j;
        float v = w[f];
        s1 += v * aw[f];
        s2 += v * aw[NF + f];
    }
#pragma unroll
    for (int off = 16; off > 0; off >>= 1) {
        s1 += __shfl_xor(s1, off, 32);
        s2 += __shfl_xor(s2, off, 32);
    }
    if (lane == 0) { f1[node] = s1; f2[node] = s2; }
}

// ---------------------------------------------------------------------------
// K3: per-edge score + LeakyReLU, segment-max via monotone-int atomicMax
// ---------------------------------------------------------------------------
__global__ void k_edge(const int* __restrict__ row, const int* __restrict__ col,
                       const float* __restrict__ f1, const float* __restrict__ f2,
                       const float* __restrict__ ab, float* __restrict__ evals,
                       int* __restrict__ menc, int E) {
    int e = blockIdx.x * blockDim.x + threadIdx.x;
    if (e >= E) return;
    float v = f1[col[e]] + f2[row[e]] + ab[0];
    v = v > 0.f ? v : 0.2f * v;                  // LeakyReLU(0.2)
    evals[e] = v;
    atomicMax(menc + row[e], encf(v));
}

// ---------------------------------------------------------------------------
// K4: one wave32 per edge: ex = exp(e - m[row]); denom[row] += ex;
//     out[row, :] += ex * Wh[col, :]   (Wh + out both L2-resident: 51MB each)
// ---------------------------------------------------------------------------
__global__ __launch_bounds__(256) void k_scatter(const int* __restrict__ row,
                                                 const int* __restrict__ col,
                                                 const float* __restrict__ evals,
                                                 const int* __restrict__ menc,
                                                 float* __restrict__ denom,
                                                 const float* __restrict__ Wh,
                                                 float* __restrict__ out, int E) {
    int gid  = blockIdx.x * blockDim.x + threadIdx.x;
    int e    = gid >> 5;
    int lane = gid & 31;
    if (e >= E) return;
    int r = row[e], c = col[e];
    float ex = __expf(evals[e] - decf(menc[r]));
    if (lane == 0) atomicAdd(denom + r, ex);
    const float* wc = Wh + (size_t)c * NF;
    float*       o  = out + (size_t)r * NF;
#pragma unroll
    for (int j = 0; j < 8; ++j) {
        int f = lane + 32 * j;                  // coalesced across the wave
        atomicAdd(o + f, ex * wc[f]);
    }
}

// ---------------------------------------------------------------------------
// K5: out = relu(out / denom[node]); empty rows -> 0
// ---------------------------------------------------------------------------
__global__ void k_finalize(float* __restrict__ out, const float* __restrict__ denom,
                           int total) {
    int idx = blockIdx.x * blockDim.x + threadIdx.x;
    if (idx >= total) return;
    float d = denom[idx >> 8];
    float v = out[idx];
    out[idx] = (d > 0.f) ? fmaxf(v / d, 0.f) : 0.f;
}

// ---------------------------------------------------------------------------
extern "C" void kernel_launch(void* const* d_in, const int* in_sizes, int n_in,
                              void* d_out, int out_size, void* d_ws, size_t ws_size,
                              hipStream_t stream) {
    const float* x   = (const float*)d_in[0];
    const float* Ww  = (const float*)d_in[1];
    const float* Wb  = (const float*)d_in[2];
    const float* aw  = (const float*)d_in[3];
    const float* ab  = (const float*)d_in[4];
    const int*   row = (const int*)d_in[5];
    const int*   col = (const int*)d_in[6];

    const int N = in_sizes[0] / NF;   // 50000
    const int E = in_sizes[5];        // 1,600,000

    auto align256 = [](size_t v) { return (v + 255) & ~(size_t)255; };
    char* ws = (char*)d_ws;
    size_t off = 0;
    float*          Wh    = (float*)(ws + off); off = align256(off + (size_t)N * NF * 4);
    unsigned short* Wt    = (unsigned short*)(ws + off);
    unsigned*       Wtu   = (unsigned*)Wt;      off = align256(off + (size_t)NF * NF * 2);
    float*          f1    = (float*)(ws + off); off = align256(off + (size_t)N * 4);
    float*          f2    = (float*)(ws + off); off = align256(off + (size_t)N * 4);
    int*            menc  = (int*)(ws + off);   off = align256(off + (size_t)N * 4);
    float*          denom = (float*)(ws + off); off = align256(off + (size_t)N * 4);
    float*          evals = (float*)(ws + off); off = align256(off + (size_t)E * 4);

    float* out = (float*)d_out;
    const int total = N * NF;

    k_init<<<(NF * NF + 255) / 256, 256, 0, stream>>>(Ww, Wt, menc, denom, N);
    k_zero<<<(total + 255) / 256, 256, 0, stream>>>(out, total);
    k_gemm<<<N / 16, 512, 0, stream>>>(x, Wtu, Wb, Wh);
    k_gemv<<<(N * 32 + 255) / 256, 256, 0, stream>>>(Wh, aw, f1, f2, N);
    k_edge<<<(E + 255) / 256, 256, 0, stream>>>(row, col, f1, f2, ab, evals, menc, E);
    k_scatter<<<(E * 32 + 255) / 256, 256, 0, stream>>>(row, col, evals, menc, denom,
                                                        Wh, out, E);
    k_finalize<<<(total + 255) / 256, 256, 0, stream>>>(out, denom, total);
}
